// MemoNet_24051816857759
// MI455X (gfx1250) — matrix operand order repaired and checked by reading, not compiled, not run on hardware
//
#include <hip/hip_runtime.h>
#include <hip/hip_bf16.h>
#include <math.h>

// ---------------- problem constants ----------------
#define BN       2048
#define MROWS    50000
#define KSEL     320
#define SSEL     20
#define TPAST    8
#define NROWS2   (BN * SSEL)      // 40960 decoder rows
#define NROWSK   (BN * KSEL)      // 655360 memkey rows
#define EPSF     1e-12f

// input indices (setup_inputs dict order, nested containers flattened in order)
#define IN_PAST   0
#define IN_ABS    1
#define IN_MEMP   2
#define IN_MEMF   3
#define IN_ENCN   4   // conv_w, conv_b, w_ih, w_hh, b_ih, b_hh  (4..9)
#define IN_ENCR   10  // (10..15)
#define IN_PQ     16  // W0,b0,W1,b1,W2,b2 (16..21)
#define IN_PMK    22  // (22..27)
#define IN_PDEC   28  // W0,b0,...,W3,b3 (28..35)
#define IN_PDECX  36  // (36..43)
#define IN_PDEC2  44  // (44..51)

typedef __attribute__((ext_vector_type(16))) _Float16 v16h;
typedef __attribute__((ext_vector_type(8)))  _Float16 v8h;
typedef __attribute__((ext_vector_type(8)))  float    v8f;

// =====================================================================
// WMMA helpers — layouts per CDNA5 ISA §7.12.2 (wave32)
// =====================================================================
// A fragment: 16x32 f16 tile, row-major source (ld halfs).
// lanes 0-15 : row = lane,    K = {k0..k0+7, k0+16..k0+23}
// lanes 16-31: row = lane-16, K = {k0+8..k0+15, k0+24..k0+31}
// Each half is an 8-half contiguous run -> two 16B ds_load_b128 per lane.
__device__ inline v16h load_A_frag(const _Float16* a, int ld, int rowBase, int k0, int lane) {
  const _Float16* p = a + (rowBase + (lane & 15)) * ld + k0 + ((lane & 16) ? 8 : 0);
  v8h lo = *(const v8h*)p;          // K = kb .. kb+7   (frag elems 0..7)
  v8h hi = *(const v8h*)(p + 16);   // K = kb+16..kb+23 (frag elems 8..15)
  return __builtin_shufflevector(lo, hi, 0, 1, 2, 3, 4, 5, 6, 7,
                                 8, 9, 10, 11, 12, 13, 14, 15);
}

// B fragment from pre-swizzled weights: tiles ordered (n-tile major, k-tile
// minor); within a tile each lane's 16-half fragment is contiguous (32 B).
// -> two coalesced global_load_b128 per lane.
__device__ inline v16h load_B_frag_swz(const _Float16* Wswz, int KT, int n0, int k0, int lane) {
  const _Float16* p = Wswz + (((size_t)(n0 >> 4) * KT + (k0 >> 5)) << 9) + (lane << 4);
  return *(const v16h*)p;
}

__device__ inline v8f wmma_f16(v16h a, v16h b, v8f c) {
  return __builtin_amdgcn_wmma_f32_16x16x32_f16(false, a, false, b, (short)0, c, false, false);
}

// One wave computes a 16-row stripe across all N output columns:
// out = act(A @ W + bias).  For K<=256, A fragments are hoisted into
// registers once and reused across all N/16 output tiles.
template <int K, int N, bool RELU, bool F16OUT>
__device__ inline void wave_layer(const _Float16* aLDS, const _Float16* Wswz,
                                  const float* bias, void* outLDS,
                                  int waveRow, int lane) {
  constexpr int KT = K / 32;
  if constexpr (KT <= 8) {
    v16h afr[KT];
#pragma unroll
    for (int kt = 0; kt < KT; ++kt)
      afr[kt] = load_A_frag(aLDS, K, waveRow, kt * 32, lane);
#pragma unroll 2
    for (int n0 = 0; n0 < N; n0 += 16) {
      v8f c = {};
#pragma unroll
      for (int kt = 0; kt < KT; ++kt) {
        v16h bf = load_B_frag_swz(Wswz, KT, n0, kt * 32, lane);
        c = wmma_f16(afr[kt], bf, c);
      }
      int col = n0 + (lane & 15);
      float bv = bias[col];
#pragma unroll
      for (int i = 0; i < 8; ++i) {
        int r = waveRow + ((lane & 16) ? (8 + i) : i);
        float v = c[i] + bv;
        if (RELU) v = fmaxf(v, 0.f);
        if (F16OUT) ((_Float16*)outLDS)[r * N + col] = (_Float16)v;
        else        ((float*)outLDS)[r * N + col] = v;
      }
    }
  } else {
    for (int n0 = 0; n0 < N; n0 += 16) {
      v8f c = {};
#pragma unroll 4
      for (int k0 = 0; k0 < K; k0 += 32) {
        v16h af = load_A_frag(aLDS, K, waveRow, k0, lane);
        v16h bf = load_B_frag_swz(Wswz, KT, n0, k0, lane);
        c = wmma_f16(af, bf, c);
      }
      int col = n0 + (lane & 15);
      float bv = bias[col];
#pragma unroll
      for (int i = 0; i < 8; ++i) {
        int r = waveRow + ((lane & 16) ? (8 + i) : i);
        float v = c[i] + bv;
        if (RELU) v = fmaxf(v, 0.f);
        if (F16OUT) ((_Float16*)outLDS)[r * N + col] = (_Float16)v;
        else        ((float*)outLDS)[r * N + col] = v;
      }
    }
  }
}

// =====================================================================
// GRU encoder (conv1d 2->16 k3 pad1 + relu, GRU 16->64), one thread/seq
// =====================================================================
struct EncLds {
  float conv_w[16 * 2 * 3];
  float conv_b[16];
  float w_ih[16 * 192];
  float w_hh[64 * 192];
  float b_ih[192];
  float b_hh[192];
};

__device__ inline void load_enc(EncLds* w, const float* cw, const float* cb,
                                const float* wih, const float* whh,
                                const float* bih, const float* bhh) {
  int tid = threadIdx.x, nt = blockDim.x;
  for (int i = tid; i < 96;    i += nt) w->conv_w[i] = cw[i];
  for (int i = tid; i < 16;    i += nt) w->conv_b[i] = cb[i];
  for (int i = tid; i < 3072;  i += nt) w->w_ih[i]  = wih[i];
  for (int i = tid; i < 12288; i += nt) w->w_hh[i]  = whh[i];
  for (int i = tid; i < 192;   i += nt) w->b_ih[i]  = bih[i];
  for (int i = tid; i < 192;   i += nt) w->b_hh[i]  = bhh[i];
}

__device__ void st_encode(const EncLds* w, const float x[TPAST][2], float h[64]) {
#pragma unroll
  for (int i = 0; i < 64; ++i) h[i] = 0.f;
  float hn[64];
  for (int t = 0; t < TPAST; ++t) {
    float xt[16];
#pragma unroll
    for (int o = 0; o < 16; ++o) {
      float acc = w->conv_b[o];
#pragma unroll
      for (int j = 0; j < 3; ++j) {
        int tt = t + j - 1;
        if (tt >= 0 && tt < TPAST) {
          acc += w->conv_w[o * 6 + j]     * x[tt][0];
          acc += w->conv_w[o * 6 + 3 + j] * x[tt][1];
        }
      }
      xt[o] = fmaxf(acc, 0.f);
    }
    for (int u = 0; u < 64; ++u) {
      float ir = w->b_ih[u], iz = w->b_ih[64 + u], in = w->b_ih[128 + u];
      for (int i = 0; i < 16; ++i) {
        float xv = xt[i]; const float* wr = &w->w_ih[i * 192];
        ir += xv * wr[u]; iz += xv * wr[64 + u]; in += xv * wr[128 + u];
      }
      float hr = w->b_hh[u], hz = w->b_hh[64 + u], hng = w->b_hh[128 + u];
      for (int i = 0; i < 64; ++i) {
        float hv = h[i]; const float* wr = &w->w_hh[i * 192];
        hr += hv * wr[u]; hz += hv * wr[64 + u]; hng += hv * wr[128 + u];
      }
      float r = 1.f / (1.f + __expf(-(ir + hr)));
      float z = 1.f / (1.f + __expf(-(iz + hz)));
      float n = tanhf(in + r * hng);
      hn[u] = (1.f - z) * n + z * h[u];
    }
#pragma unroll
    for (int u = 0; u < 64; ++u) h[u] = hn[u];
  }
}

// =====================================================================
// K0: weight f32->f16 conversion into WMMA fragment-major (swizzled)
// layout with optional N padding.  Output element i decomposes as:
//   tile = i/512 (n-major: tile = (n0/16)*KT + k0/32), lane = (i%512)/16,
//   e = i%16;  element (lane,e) holds W[k0 + laneHi*8 + (e&7) + (e&8 ? 16 : 0)][n0 + lane&15].
// =====================================================================
__global__ void cvt_w_swz(const float* __restrict__ src, _Float16* __restrict__ dst,
                          int K, int N, int Npad) {
  int i = blockIdx.x * blockDim.x + threadIdx.x;
  if (i >= K * Npad) return;
  int KT = K >> 5;
  int tile = i >> 9;
  int k0 = (tile % KT) << 5;
  int n0 = (tile / KT) << 4;
  int wi = i & 511;
  int lane = wi >> 4, e = wi & 15;
  int c = n0 + (lane & 15);
  int k = k0 + ((lane & 16) ? 8 : 0) + (e & 7) + ((e & 8) ? 16 : 0);
  dst[i] = (_Float16)((c < N) ? src[k * N + c] : 0.f);
}

__global__ void pad_bias(const float* __restrict__ src, float* __restrict__ dst,
                         int n, int npad) {
  int i = threadIdx.x;
  if (i < npad) dst[i] = (i < n) ? src[i] : 0.f;
}

// =====================================================================
// K1: encode past -> state_past [B,128], nq (l2-normalized) [B,128]
// =====================================================================
__global__ void encode_past_kernel(const float* __restrict__ past,
                                   const float* __restrict__ abss,
                                   const float* cw, const float* cb, const float* wih,
                                   const float* whh, const float* bih, const float* bhh,
                                   float* __restrict__ state_past,
                                   float* __restrict__ nq) {
  extern __shared__ char smem[];
  EncLds* w = (EncLds*)smem;
  load_enc(w, cw, cb, wih, whh, bih, bhh);
  __syncthreads();
  int b = blockIdx.x * blockDim.x + threadIdx.x;
  if (b >= BN) return;
  float x[TPAST][2];
#pragma unroll
  for (int t = 0; t < TPAST; ++t) { x[t][0] = past[b * 16 + t * 2]; x[t][1] = past[b * 16 + t * 2 + 1]; }
  float h[64];
  st_encode(w, x, h);
  float ss = 0.f;
  float sp[128];
#pragma unroll
  for (int i = 0; i < 64; ++i) sp[i] = h[i];
#pragma unroll
  for (int i = 0; i < 64; ++i) sp[64 + i] = abss[b * 64 + i];
#pragma unroll
  for (int i = 0; i < 128; ++i) ss += sp[i] * sp[i];
  float rinv = 1.f / fmaxf(sqrtf(ss), EPSF);
#pragma unroll
  for (int i = 0; i < 128; ++i) {
    state_past[b * 128 + i] = sp[i];
    nq[b * 128 + i] = sp[i] * rinv;
  }
}

// =====================================================================
// K2: reciprocal row norms of memory_past
// =====================================================================
__global__ void mem_rnorm_kernel(const float* __restrict__ memp, float* __restrict__ rnorm) {
  int m = blockIdx.x * blockDim.x + threadIdx.x;
  if (m >= MROWS) return;
  const float* r = memp + (size_t)m * 128;
  float ss = 0.f;
#pragma unroll 8
  for (int j = 0; j < 128; ++j) ss += r[j] * r[j];
  rnorm[m] = 1.f / fmaxf(sqrtf(ss), EPSF);
}

// =====================================================================
// K3: cosine sim + top-320 per row via 12-bit radix histogram threshold
// one block (256 threads) per batch row; sims recomputed in pass 2
// =====================================================================
__global__ void sim_topk_kernel(const float* __restrict__ nq,
                                const float* __restrict__ memp,
                                const float* __restrict__ rnorm,
                                int* __restrict__ idx320) {
  __shared__ float qv[128];
  __shared__ unsigned hist[4096];
  __shared__ int s_thr, s_chi, cnt_hi, cnt_tie;
  int b = blockIdx.x, tid = threadIdx.x;
  for (int i = tid; i < 128; i += 256) qv[i] = nq[b * 128 + i];
  for (int i = tid; i < 4096; i += 256) hist[i] = 0u;
  __syncthreads();
  for (int m = tid; m < MROWS; m += 256) {
    const float* mr = memp + (size_t)m * 128;
    float s = 0.f;
#pragma unroll 8
    for (int j = 0; j < 128; ++j) s += qv[j] * mr[j];
    s *= rnorm[m];
    unsigned u = __float_as_uint(s);
    u = (u & 0x80000000u) ? ~u : (u | 0x80000000u);
    atomicAdd(&hist[u >> 20], 1u);
  }
  __syncthreads();
  if (tid == 0) {
    int c = 0, t = 0;
    for (int bn = 4095; bn >= 0; --bn) {
      int nc = c + (int)hist[bn];
      if (nc >= KSEL) { t = bn; break; }
      c = nc;
    }
    s_thr = t; s_chi = c; cnt_hi = 0; cnt_tie = 0;
  }
  __syncthreads();
  int thr = s_thr, chi = s_chi;
  for (int m = tid; m < MROWS; m += 256) {
    const float* mr = memp + (size_t)m * 128;
    float s = 0.f;
#pragma unroll 8
    for (int j = 0; j < 128; ++j) s += qv[j] * mr[j];
    s *= rnorm[m];
    unsigned u = __float_as_uint(s);
    u = (u & 0x80000000u) ? ~u : (u | 0x80000000u);
    int bn = (int)(u >> 20);
    if (bn > thr) {
      int p = atomicAdd(&cnt_hi, 1);
      idx320[b * KSEL + p] = m;
    } else if (bn == thr) {
      int p = atomicAdd(&cnt_tie, 1);
      if (chi + p < KSEL) idx320[b * KSEL + chi + p] = m;
    }
  }
}

// =====================================================================
// K4: fused 3-layer MLP (128->256->256->128) with WMMA.
// mode 0: query  (input = state_past rows, output = l2norm -> qproj)
// mode 1: memkey (input = gather memory_past[idx320], output = w2)
// block = 128 threads (4 waves), tile = 64 rows
// =====================================================================
__global__ void mlp3_kernel(int mode,
                            const float* __restrict__ srcRows,
                            const float* __restrict__ memp,
                            const int* __restrict__ idx320,
                            const _Float16* W1, const float* b1,
                            const _Float16* W2, const float* b2,
                            const _Float16* W3, const float* b3,
                            float* __restrict__ qproj,
                            float* __restrict__ w2out) {
  extern __shared__ char smem[];
  _Float16* actA = (_Float16*)smem;              // 64*256 f16
  _Float16* actB = actA + 64 * 256;              // 64*256 f16
  float*    actC = (float*)(actB + 64 * 256);    // 64*128 f32
  int t0 = blockIdx.x * 64;
  int tid = threadIdx.x, lane = tid & 31, waveRow = (tid >> 5) * 16;

  for (int e = tid; e < 64 * 128; e += 128) {
    int r = e >> 7, j = e & 127, row = t0 + r;
    float v = (mode == 0) ? srcRows[(size_t)row * 128 + j]
                          : memp[(size_t)idx320[row] * 128 + j];
    actA[r * 128 + j] = (_Float16)v;
  }
  __syncthreads();
  wave_layer<128, 256, true,  true >(actA, W1, b1, actB, waveRow, lane);
  __syncthreads();
  wave_layer<256, 256, true,  true >(actB, W2, b2, actA, waveRow, lane);
  __syncthreads();
  wave_layer<256, 128, false, false>(actA, W3, b3, actC, waveRow, lane);
  __syncthreads();
  if (tid < 64) {
    int row = t0 + tid;
    const float* o = actC + tid * 128;
    float ss = 0.f;
#pragma unroll 8
    for (int j = 0; j < 128; ++j) ss += o[j] * o[j];
    float rinv = 1.f / fmaxf(sqrtf(ss), EPSF);
    if (mode == 0) {
      for (int j = 0; j < 128; ++j) qproj[(size_t)row * 128 + j] = o[j] * rinv;
    } else {
      int b = row / KSEL;
      const float* q = qproj + (size_t)b * 128;
      float d = 0.f;
#pragma unroll 8
      for (int j = 0; j < 128; ++j) d += o[j] * q[j];
      w2out[row] = d * rinv;
    }
  }
}

// =====================================================================
// K5: top-20 of w2 per row (iterated argmax), build state_conc f16
// =====================================================================
__global__ void select20_kernel(const float* __restrict__ w2,
                                const int* __restrict__ idx320,
                                const float* __restrict__ state_past,
                                const float* __restrict__ memf,
                                _Float16* __restrict__ state_conc) {
  __shared__ float vals[KSEL];
  __shared__ int   msel[SSEL];
  __shared__ float rv[256];
  __shared__ int   ri[256];
  int b = blockIdx.x, tid = threadIdx.x;
  for (int k = tid; k < KSEL; k += 256) vals[k] = w2[b * KSEL + k];
  __syncthreads();
  for (int s = 0; s < SSEL; ++s) {
    float best = -1e30f; int bi = KSEL;
    for (int k = tid; k < KSEL; k += 256) {
      float v = vals[k];
      if (v > best) { best = v; bi = k; }
    }
    rv[tid] = best; ri[tid] = bi;
    __syncthreads();
    for (int o = 128; o > 0; o >>= 1) {
      if (tid < o) {
        if (rv[tid + o] > rv[tid] ||
            (rv[tid + o] == rv[tid] && ri[tid + o] < ri[tid])) {
          rv[tid] = rv[tid + o]; ri[tid] = ri[tid + o];
        }
      }
      __syncthreads();
    }
    if (tid == 0) { msel[s] = idx320[b * KSEL + ri[0]]; vals[ri[0]] = -1e30f; }
    __syncthreads();
  }
  // state_conc rows: [state_past(128) | memory_fut[sel](64)]
  for (int e = tid; e < SSEL * 128; e += 256) {
    int s = e >> 7, j = e & 127;
    state_conc[((size_t)(b * SSEL + s)) * 192 + j] = (_Float16)state_past[b * 128 + j];
  }
  for (int e = tid; e < SSEL * 64; e += 256) {
    int s = e >> 6, j = e & 63;
    state_conc[((size_t)(b * SSEL + s)) * 192 + 128 + j] =
        (_Float16)memf[(size_t)msel[s] * 64 + j];
  }
}

// =====================================================================
// K6: fused decoder MLP 192->1024->512->1024->OUT (WMMA), 64-row tiles
// =====================================================================
__global__ void dec_mlp_kernel(const _Float16* __restrict__ actIn,
                               const _Float16* W1, const float* b1,
                               const _Float16* W2, const float* b2,
                               const _Float16* W3, const float* b3,
                               const _Float16* W4, const float* b4pad,
                               int outReal,
                               float* __restrict__ outG,
                               const float* __restrict__ addend) {
  extern __shared__ char smem[];
  _Float16* bufA = (_Float16*)smem;            // 64*1024 f16
  _Float16* bufB = bufA + 64 * 1024;           // 64*512  f16
  _Float16* bufI = bufB + 64 * 512;            // 64*192  f16
  int t0 = blockIdx.x * 64;
  int tid = threadIdx.x, lane = tid & 31, waveRow = (tid >> 5) * 16;

  for (int e = tid; e < 64 * 192; e += 128) bufI[e] = actIn[(size_t)t0 * 192 + e];
  __syncthreads();
  wave_layer<192,  1024, true, true>(bufI, W1, b1, bufA, waveRow, lane);
  __syncthreads();
  wave_layer<1024, 512,  true, true>(bufA, W2, b2, bufB, waveRow, lane);
  __syncthreads();
  wave_layer<512,  1024, true, true>(bufB, W3, b3, bufA, waveRow, lane);
  __syncthreads();
  // final layer: single 16-wide tile (padded weights), no relu, f32 global
  v8f c = {};
#pragma unroll 4
  for (int k0 = 0; k0 < 1024; k0 += 32) {
    v16h af = load_A_frag(bufA, 1024, waveRow, k0, lane);
    v16h bf = load_B_frag_swz(W4, 32, 0, k0, lane);
    c = wmma_f16(af, bf, c);
  }
  int col = lane & 15;
  if (col < outReal) {
    float bv = b4pad[col];
#pragma unroll
    for (int i = 0; i < 8; ++i) {
      int r = waveRow + ((lane & 16) ? (8 + i) : i);
      int row = t0 + r;
      float v = c[i] + bv;
      if (addend) v += addend[(size_t)row * outReal + col];
      outG[(size_t)row * outReal + col] = v;
    }
  }
}

// =====================================================================
// K7: residual (diff) GRU encoder + build state_conc_diff f16
// =====================================================================
__global__ void diff_encode_kernel(const float* __restrict__ past,
                                   const float* __restrict__ rx1,
                                   const float* __restrict__ abss,
                                   const _Float16* __restrict__ state_conc,
                                   const float* cw, const float* cb, const float* wih,
                                   const float* whh, const float* bih, const float* bhh,
                                   _Float16* __restrict__ state_conc_diff) {
  extern __shared__ char smem[];
  EncLds* w = (EncLds*)smem;
  load_enc(w, cw, cb, wih, whh, bih, bhh);
  __syncthreads();
  int t = blockIdx.x * blockDim.x + threadIdx.x;
  if (t >= NROWS2) return;
  int b = t / SSEL;
  float x[TPAST][2];
#pragma unroll
  for (int j = 0; j < TPAST; ++j) {
    x[j][0] = past[b * 16 + j * 2]     - rx1[(size_t)t * 16 + j * 2];
    x[j][1] = past[b * 16 + j * 2 + 1] - rx1[(size_t)t * 16 + j * 2 + 1];
  }
  float h[64];
  st_encode(w, x, h);
  _Float16* o = state_conc_diff + (size_t)t * 192;
#pragma unroll
  for (int u = 0; u < 64; ++u) o[u] = (_Float16)h[u];
#pragma unroll
  for (int u = 0; u < 64; ++u) o[64 + u] = (_Float16)abss[b * 64 + u];
#pragma unroll
  for (int u = 0; u < 64; ++u) o[128 + u] = state_conc[(size_t)t * 192 + 128 + u];
}

// =====================================================================
// host side
// =====================================================================
static inline size_t alignup(size_t x) { return (x + 255) & ~(size_t)255; }

extern "C" void kernel_launch(void* const* d_in, const int* in_sizes, int n_in,
                              void* d_out, int out_size, void* d_ws, size_t ws_size,
                              hipStream_t stream) {
  (void)in_sizes; (void)n_in; (void)out_size; (void)ws_size;
  const float* past = (const float*)d_in[IN_PAST];
  const float* abss = (const float*)d_in[IN_ABS];
  const float* memp = (const float*)d_in[IN_MEMP];
  const float* memf = (const float*)d_in[IN_MEMF];
  float* out = (float*)d_out;

  // ---- carve workspace ----
  char* ws = (char*)d_ws; size_t off = 0;
  auto carve = [&](size_t bytes) { void* p = ws + off; off += alignup(bytes); return p; };
  float* state_past = (float*)carve((size_t)BN * 128 * 4);
  float* nq         = (float*)carve((size_t)BN * 128 * 4);
  float* qproj      = (float*)carve((size_t)BN * 128 * 4);
  float* rnorm      = (float*)carve((size_t)MROWS * 4);
  int*   idx320     = (int*)  carve((size_t)NROWSK * 4);
  float* w2buf      = (float*)carve((size_t)NROWSK * 4);
  _Float16* state_conc      = (_Float16*)carve((size_t)NROWS2 * 192 * 2);
  _Float16* state_conc_diff = (_Float16*)carve((size_t)NROWS2 * 192 * 2);
  float* y1  = (float*)carve((size_t)NROWS2 * 2 * 4);
  float* rx1 = (float*)carve((size_t)NROWS2 * 16 * 4);

  // f16 weights (swizzled, padded)
  _Float16* wq[3]; _Float16* wm[3];
  const int qdims[4] = {128, 256, 256, 128};
  for (int l = 0; l < 3; ++l) {
    wq[l] = (_Float16*)carve((size_t)qdims[l] * qdims[l + 1] * 2);
    wm[l] = (_Float16*)carve((size_t)qdims[l] * qdims[l + 1] * 2);
  }
  const int ddims[5] = {192, 1024, 512, 1024, 16};   // last padded to 16
  _Float16 *wd[4], *wx[4], *w2d[4];
  for (int l = 0; l < 4; ++l) {
    wd[l]  = (_Float16*)carve((size_t)ddims[l] * ddims[l + 1] * 2);
    wx[l]  = (_Float16*)carve((size_t)ddims[l] * ddims[l + 1] * 2);
    w2d[l] = (_Float16*)carve((size_t)ddims[l] * ddims[l + 1] * 2);
  }
  float* bd4 = (float*)carve(16 * 4);
  float* bx4 = (float*)carve(16 * 4);
  float* b24 = (float*)carve(16 * 4);

  // ---- K0: weight conversion (swizzle to fragment-major) ----
  auto cvt = [&](const float* src, _Float16* dst, int K, int N, int Npad) {
    int total = K * Npad;
    cvt_w_swz<<<(total + 255) / 256, 256, 0, stream>>>(src, dst, K, N, Npad);
  };
  for (int l = 0; l < 3; ++l) {
    cvt((const float*)d_in[IN_PQ  + 2 * l], wq[l], qdims[l], qdims[l + 1], qdims[l + 1]);
    cvt((const float*)d_in[IN_PMK + 2 * l], wm[l], qdims[l], qdims[l + 1], qdims[l + 1]);
  }
  const int drealN[4]  = {1024, 512, 1024, 2};
  const int dxrealN[4] = {1024, 512, 1024, 16};
  for (int l = 0; l < 4; ++l) {
    cvt((const float*)d_in[IN_PDEC  + 2 * l], wd[l],  ddims[l], drealN[l],  ddims[l + 1]);
    cvt((const float*)d_in[IN_PDECX + 2 * l], wx[l],  ddims[l], dxrealN[l], ddims[l + 1]);
    cvt((const float*)d_in[IN_PDEC2 + 2 * l], w2d[l], ddims[l], drealN[l],  ddims[l + 1]);
  }
  pad_bias<<<1, 16, 0, stream>>>((const float*)d_in[IN_PDEC  + 7], bd4, 2, 16);
  pad_bias<<<1, 16, 0, stream>>>((const float*)d_in[IN_PDECX + 7], bx4, 16, 16);
  pad_bias<<<1, 16, 0, stream>>>((const float*)d_in[IN_PDEC2 + 7], b24, 2, 16);

  size_t encLds = sizeof(EncLds);

  // ---- K1: encode past ----
  encode_past_kernel<<<BN / 256, 256, encLds, stream>>>(
      past, abss,
      (const float*)d_in[IN_ENCN], (const float*)d_in[IN_ENCN + 1],
      (const float*)d_in[IN_ENCN + 2], (const float*)d_in[IN_ENCN + 3],
      (const float*)d_in[IN_ENCN + 4], (const float*)d_in[IN_ENCN + 5],
      state_past, nq);

  // ---- K2: memory row norms ----
  mem_rnorm_kernel<<<(MROWS + 255) / 256, 256, 0, stream>>>(memp, rnorm);

  // ---- K3: cosine sim + top-320 ----
  sim_topk_kernel<<<BN, 256, 0, stream>>>(nq, memp, rnorm, idx320);

  // ---- K4: query projection then memkey + w2 (WMMA) ----
  size_t mlp3Lds = (size_t)64 * 256 * 2 * 2 + (size_t)64 * 128 * 4;   // 98304 B
  mlp3_kernel<<<BN / 64, 128, mlp3Lds, stream>>>(
      0, state_past, memp, idx320,
      wq[0], (const float*)d_in[IN_PQ + 1],
      wq[1], (const float*)d_in[IN_PQ + 3],
      wq[2], (const float*)d_in[IN_PQ + 5],
      qproj, w2buf);
  mlp3_kernel<<<NROWSK / 64, 128, mlp3Lds, stream>>>(
      1, state_past, memp, idx320,
      wm[0], (const float*)d_in[IN_PMK + 1],
      wm[1], (const float*)d_in[IN_PMK + 3],
      wm[2], (const float*)d_in[IN_PMK + 5],
      qproj, w2buf);

  // ---- K5: top-20 + state_conc ----
  select20_kernel<<<BN, 256, 0, stream>>>(w2buf, idx320, state_past, memf, state_conc);

  // ---- K6: decoder MLPs (WMMA) ----
  size_t decLds = (size_t)64 * 1024 * 2 + (size_t)64 * 512 * 2 + (size_t)64 * 192 * 2; // 221184 B
  dec_mlp_kernel<<<NROWS2 / 64, 128, decLds, stream>>>(
      state_conc,
      wd[0], (const float*)d_in[IN_PDEC + 1],
      wd[1], (const float*)d_in[IN_PDEC + 3],
      wd[2], (const float*)d_in[IN_PDEC + 5],
      wd[3], bd4, 2, y1, nullptr);
  dec_mlp_kernel<<<NROWS2 / 64, 128, decLds, stream>>>(
      state_conc,
      wx[0], (const float*)d_in[IN_PDECX + 1],
      wx[1], (const float*)d_in[IN_PDECX + 3],
      wx[2], (const float*)d_in[IN_PDECX + 5],
      wx[3], bx4, 16, rx1, nullptr);

  // ---- K7: residual encoder ----
  diff_encode_kernel<<<NROWS2 / 256, 256, encLds, stream>>>(
      past, rx1, abss, state_conc,
      (const float*)d_in[IN_ENCR], (const float*)d_in[IN_ENCR + 1],
      (const float*)d_in[IN_ENCR + 2], (const float*)d_in[IN_ENCR + 3],
      (const float*)d_in[IN_ENCR + 4], (const float*)d_in[IN_ENCR + 5],
      state_conc_diff);

  // ---- K6 again: dec2, fused add of y1, writes final output ----
  dec_mlp_kernel<<<NROWS2 / 64, 128, decLds, stream>>>(
      state_conc_diff,
      w2d[0], (const float*)d_in[IN_PDEC2 + 1],
      w2d[1], (const float*)d_in[IN_PDEC2 + 3],
      w2d[2], (const float*)d_in[IN_PDEC2 + 5],
      w2d[3], b24, 2, out, y1);
}
